// MultiHeadedSelfAttention_80324478369926
// MI455X (gfx1250) — matrix operand order repaired and checked by reading, not compiled, not run on hardware
//
#include <hip/hip_runtime.h>

// ---------------------------------------------------------------------------
// MultiHeadedSelfAttention for MI455X (gfx1250), f16 WMMA + flash attention.
// B=4, S=2048, E=1024, H=16, DH=64. Inputs f32, output f32.
// - All LDS staging via gfx1250 async global->LDS copies, double-buffered via
//   SWAPPED BUFFER POINTERS (not parity indexing) so the loop body is
//   iteration-invariant: no unroll-by-2, accumulators stay pinned, no
//   v_mov_b64 shuffles or WMMA->VALU hazard nops.
// - All WMMA fragments are laid out as exactly two ds_load_b128 each.
// ---------------------------------------------------------------------------

#define B_  4
#define S_  2048
#define E_  1024
#define H_  16
#define DH_ 64

typedef __attribute__((ext_vector_type(16))) _Float16 v16h;
typedef __attribute__((ext_vector_type(8)))  float    v8f;

union HFrag { v16h v; unsigned u[8]; uint4 q[2]; };

__device__ __forceinline__ unsigned short f16b(float x) {
    union { _Float16 h; unsigned short u; } c;
    c.h = (_Float16)x;
    return c.u;
}

// gfx1250 async copy: 16B global -> LDS, tracked by ASYNCcnt.
__device__ __forceinline__ void async_b128(void* lds_ptr, const void* g_ptr) {
    unsigned lds_off = (unsigned)(unsigned long long)lds_ptr;   // low 32b of flat = LDS offset
    unsigned long long ga = (unsigned long long)g_ptr;
    asm volatile("global_load_async_to_lds_b128 %0, %1, off"
                 :: "v"(lds_off), "v"(ga) : "memory");
}
__device__ __forceinline__ void wait_async0() { asm volatile("s_wait_asynccnt 0" ::: "memory"); }
__device__ __forceinline__ void wait_async4() { asm volatile("s_wait_asynccnt 4" ::: "memory"); }
__device__ __forceinline__ void wait_async5() { asm volatile("s_wait_asynccnt 5" ::: "memory"); }

// Row reductions across the 16 lanes holding one C/D-matrix row.
__device__ __forceinline__ float rowred_max(float x) {
    x = fmaxf(x, __shfl_xor(x, 1, 32));
    x = fmaxf(x, __shfl_xor(x, 2, 32));
    x = fmaxf(x, __shfl_xor(x, 4, 32));
    x = fmaxf(x, __shfl_xor(x, 8, 32));
    return x;
}
__device__ __forceinline__ float rowred_sum(float x) {
    x += __shfl_xor(x, 1, 32);
    x += __shfl_xor(x, 2, 32);
    x += __shfl_xor(x, 4, 32);
    x += __shfl_xor(x, 8, 32);
    return x;
}

// ---------------------------------------------------------------------------
// Kernel 1: f32 -> f16 conversion (activations X)
// ---------------------------------------------------------------------------
__global__ __launch_bounds__(256)
void cvt_f16_kernel(const float* __restrict__ src, unsigned short* __restrict__ dst, int n4) {
    int i = blockIdx.x * blockDim.x + threadIdx.x;
    int stride = gridDim.x * blockDim.x;
    for (; i < n4; i += stride) {
        float4 f = ((const float4*)src)[i];
        ushort4 h;
        h.x = f16b(f.x); h.y = f16b(f.y); h.z = f16b(f.z); h.w = f16b(f.w);
        ((ushort4*)dst)[i] = h;
    }
}

// ---------------------------------------------------------------------------
// Kernel 2: weight swizzle: W (H,E,DH) f32 -> fragment-major f16.
// Per (h, e-block of 32): 64 cols x 32 contiguous k -> proj B tile is a
// straight contiguous 4KB async copy into LDS.
// ---------------------------------------------------------------------------
__global__ __launch_bounds__(256)
void wswz_kernel(const float* __restrict__ W, unsigned short* __restrict__ dst) {
    int t = blockIdx.x * blockDim.x + threadIdx.x;   // over H*E*DH
    int c = t & 63;
    int e = (t >> 6) & (E_ - 1);
    int h = t >> 16;
    float val = W[t];
    dst[((h * 32 + (e >> 5)) * 64 + c) * 32 + (e & 31)] = f16b(val);
}

// ---------------------------------------------------------------------------
// Kernel 3: QKV projection, double-buffered (pointer-swap).
//   grid = (B*H, S/256, 3[q,k,v]), block = 256 (8 waves).
//   Each wave computes 32x64 (2 A-frags x 4 B-frags = 8 WMMA per K-step).
//   Q,K stored (B,H,S,DH); V stored transposed (B,H,DH,S).
// ---------------------------------------------------------------------------
__global__ __launch_bounds__(256)
void qkv_proj_kernel(const unsigned short* __restrict__ Xh,    // (B,S,E)
                     const unsigned short* __restrict__ Wswz,  // (3,H,32,64,32)
                     unsigned short* __restrict__ Qh,          // (B,H,S,DH)
                     unsigned short* __restrict__ Kh,          // (B,H,S,DH)
                     unsigned short* __restrict__ Vt)          // (B,H,DH,S)
{
    const int bh = blockIdx.x;
    const int b  = bh >> 4, h = bh & 15;
    const int s0 = blockIdx.y * 256;
    const int z  = blockIdx.z;

    __shared__ __align__(16) unsigned short ldsA[2][256 * 32];  // 2 x 16 KB (row, k)
    __shared__ __align__(16) unsigned short ldsB[2][64 * 32];   // 2 x  4 KB frag-major [c][k]

    const int tid  = threadIdx.x;
    const int wave = tid >> 5;
    const int lane = tid & 31;
    const int ln16 = lane & 15;
    const int lhi  = lane >> 4;

    v8f acc[2][4];
#pragma unroll
    for (int g2 = 0; g2 < 2; ++g2)
#pragma unroll
        for (int t = 0; t < 4; ++t)
#pragma unroll
            for (int i = 0; i < 8; ++i) acc[g2][t][i] = 0.0f;

    const unsigned short* Xbase = Xh + (size_t)(b * S_ + s0) * E_;
    const unsigned short* Wbase = Wswz + (size_t)(z * H_ + h) * (32 * 2048);

    // 5 async ops per thread per tile (per-wave ASYNCcnt += 5)
    auto issue_tile = [&](int e0, unsigned short* bufA, unsigned short* bufB) {
        const unsigned short* g = Xbase + (size_t)tid * E_ + e0;
        unsigned short* l = bufA + tid * 32;
        async_b128(l,      g);
        async_b128(l + 8,  g + 8);
        async_b128(l + 16, g + 16);
        async_b128(l + 24, g + 24);
        async_b128(bufB + tid * 8, Wbase + (e0 >> 5) * 2048 + tid * 8);
    };

    unsigned short* aCur = &ldsA[0][0]; unsigned short* aNxt = &ldsA[1][0];
    unsigned short* bCur = &ldsB[0][0]; unsigned short* bNxt = &ldsB[1][0];

    issue_tile(0, aCur, bCur);

#pragma clang loop unroll(disable)
    for (int step = 0; step < E_ / 32; ++step) {
        __syncthreads();                       // prev compute done -> next buffer free
        if (step + 1 < E_ / 32) {
            issue_tile((step + 1) * 32, aNxt, bNxt);
            wait_async5();                     // current tile's 5 ops retired
        } else {
            wait_async0();
        }
        __syncthreads();                       // current tile visible to all waves

        const unsigned* lA = (const unsigned*)aCur;
        const unsigned* lB = (const unsigned*)bCur;

        HFrag af[2];
#pragma unroll
        for (int g2 = 0; g2 < 2; ++g2) {
            int arow = wave * 32 + g2 * 16 + ln16;
            af[g2].q[0] = *(const uint4*)(lA + arow * 16 + lhi * 4);
            af[g2].q[1] = *(const uint4*)(lA + arow * 16 + 8 + lhi * 4);
        }
#pragma unroll
        for (int t = 0; t < 4; ++t) {
            HFrag bf;
            int c = t * 16 + ln16;
            bf.q[0] = *(const uint4*)(lB + c * 16 + lhi * 4);
            bf.q[1] = *(const uint4*)(lB + c * 16 + 8 + lhi * 4);
            acc[0][t] = __builtin_amdgcn_wmma_f32_16x16x32_f16(
                false, af[0].v, false, bf.v, (short)0, acc[0][t], false, false);
            acc[1][t] = __builtin_amdgcn_wmma_f32_16x16x32_f16(
                false, af[1].v, false, bf.v, (short)0, acc[1][t], false, false);
        }

        unsigned short* t0 = aCur; aCur = aNxt; aNxt = t0;
        unsigned short* t1 = bCur; bCur = bNxt; bNxt = t1;
    }

    // C/D layout: VGPR v holds row v (+8 for lanes 16-31), col = lane%16.
    if (z == 2) {   // V transposed: (B,H,DH,S)
        unsigned short* Out = Vt + (size_t)bh * DH_ * S_;
#pragma unroll
        for (int g2 = 0; g2 < 2; ++g2)
#pragma unroll
            for (int t = 0; t < 4; ++t)
#pragma unroll
                for (int v = 0; v < 8; ++v) {
                    int r = s0 + wave * 32 + g2 * 16 + v + 8 * lhi;
                    int d = t * 16 + ln16;
                    Out[(size_t)d * S_ + r] = f16b(acc[g2][t][v]);
                }
    } else {        // Q or K: (B,H,S,DH)
        unsigned short* Out = (z == 0 ? Qh : Kh) + (size_t)bh * S_ * DH_;
#pragma unroll
        for (int g2 = 0; g2 < 2; ++g2)
#pragma unroll
            for (int t = 0; t < 4; ++t)
#pragma unroll
                for (int v = 0; v < 8; ++v) {
                    int r = s0 + wave * 32 + g2 * 16 + v + 8 * lhi;
                    Out[(size_t)r * DH_ + t * 16 + ln16] = f16b(acc[g2][t][v]);
                }
    }
}

// ---------------------------------------------------------------------------
// Kernel 4: causal flash attention, 64-wide k-chunks, double-buffered
// (pointer-swap).  grid = (B*H, S/128), block = 256 (8 waves, 16 q-rows each).
// 16 WMMAs per chunk (8 scores + 8 P.V) per 2 barriers.
// ---------------------------------------------------------------------------
__global__ __launch_bounds__(256)
void flash_attn_kernel(const unsigned short* __restrict__ Qh,  // (B,H,S,DH)
                       const unsigned short* __restrict__ Kh,  // (B,H,S,DH)
                       const unsigned short* __restrict__ Vt,  // (B,H,DH,S)
                       float* __restrict__ Out)                // (B,S,H*DH)
{
    const int bh = blockIdx.x;
    const int b  = bh >> 4, h = bh & 15;
    const int q0 = blockIdx.y * 128;

    __shared__ __align__(16) unsigned short ldsQ[128 * 64];      // 16 KB (q,d)
    __shared__ __align__(16) unsigned short ldsKc[2][64 * 64];   // 2 x 8 KB (s,d)
    __shared__ __align__(16) unsigned short ldsVt[2][64 * 64];   // 2 x 8 KB (d,k)
    __shared__ __align__(16) unsigned short ldsP[8][16 * 64];    // 16 KB per-wave P strips

    const int tid  = threadIdx.x;
    const int wave = tid >> 5;
    const int lane = tid & 31;
    const int ln16 = lane & 15;
    const int lhi  = lane >> 4;

    const unsigned short* KB  = Kh + (size_t)bh * S_ * DH_;
    const unsigned short* VtB = Vt + (size_t)bh * DH_ * S_;

    // 4 async ops per thread per chunk (per-wave ASYNCcnt += 4)
    auto issue_chunk = [&](int kc, unsigned short* bufK, unsigned short* bufV) {
        int r  = tid >> 2;            // 0..63
        int c0 = (tid & 3) * 16;      // 0..48
        const unsigned short* gk = KB + (size_t)(kc + r) * DH_ + c0;
        unsigned short* lk = bufK + r * 64 + c0;
        async_b128(lk,     gk);
        async_b128(lk + 8, gk + 8);
        const unsigned short* gv = VtB + (size_t)r * S_ + kc + c0;
        unsigned short* lv = bufV + r * 64 + c0;
        async_b128(lv,     gv);
        async_b128(lv + 8, gv + 8);
    };

    unsigned short* kCur = &ldsKc[0][0]; unsigned short* kNxt = &ldsKc[1][0];
    unsigned short* vCur = &ldsVt[0][0]; unsigned short* vNxt = &ldsVt[1][0];

    {   // Q tile: 128 x 64, 64B per thread (4 async b128)
        int row = tid >> 1;
        int off = (tid & 1) * 32;
        const unsigned short* g = Qh + ((size_t)bh * S_ + q0 + row) * DH_ + off;
        unsigned short* l = ldsQ + row * 64 + off;
        async_b128(l,      g);
        async_b128(l + 8,  g + 8);
        async_b128(l + 16, g + 16);
        async_b128(l + 24, g + 24);
    }
    issue_chunk(0, kCur, vCur);  // prefetch first K/V chunk behind the Q copy
    wait_async4();               // Q's 4 ops retired (chunk 0 may still be in flight)
    __syncthreads();

    // Q A-fragments, kept in VGPRs for the whole kernel (g = d 0-31 / 32-63)
    HFrag qf[2];
    {
        const unsigned* lQ = (const unsigned*)ldsQ;
        int qrow = wave * 16 + ln16;
#pragma unroll
        for (int g = 0; g < 2; ++g) {
            qf[g].q[0] = *(const uint4*)(lQ + qrow * 32 + g * 16 + lhi * 4);
            qf[g].q[1] = *(const uint4*)(lQ + qrow * 32 + g * 16 + 8 + lhi * 4);
        }
    }

    v8f o[4];
    float m8[8], l8[8];
#pragma unroll
    for (int t = 0; t < 4; ++t)
#pragma unroll
        for (int i = 0; i < 8; ++i) o[t][i] = 0.0f;
#pragma unroll
    for (int v = 0; v < 8; ++v) { m8[v] = -INFINITY; l8[v] = 0.0f; }

    const int qw_end = q0 + wave * 16 + 15;
    const int nch    = (q0 + 128) / 64;
    const unsigned* lP = (const unsigned*)&ldsP[wave][0];

#pragma clang loop unroll(disable)
    for (int ci = 0; ci < nch; ++ci) {
        const int kc = ci * 64;
        __syncthreads();                    // prev compute done -> next buffer free
        if (ci + 1 < nch) {
            issue_chunk(kc + 64, kNxt, vNxt);
            wait_async4();                  // current chunk's 4 ops retired
        } else {
            wait_async0();
        }
        __syncthreads();                    // current chunk visible to all waves

        if (kc <= qw_end) {                 // else fully masked for this wave (uniform)
            const unsigned* lK = (const unsigned*)kCur;
            const unsigned* lV = (const unsigned*)vCur;

            // ---- scores: four 16x16 n-tiles, contraction over d = 64 ----
            v8f st[4];
#pragma unroll
            for (int t = 0; t < 4; ++t) {
                v8f a;
#pragma unroll
                for (int i = 0; i < 8; ++i) a[i] = 0.0f;
                int n = t * 16 + ln16;
#pragma unroll
                for (int g = 0; g < 2; ++g) {
                    HFrag bf;
                    bf.q[0] = *(const uint4*)(lK + n * 32 + g * 16 + lhi * 4);
                    bf.q[1] = *(const uint4*)(lK + n * 32 + g * 16 + 8 + lhi * 4);
                    a = __builtin_amdgcn_wmma_f32_16x16x32_f16(
                        false, qf[g].v, false, bf.v, (short)0, a, false, false);
                }
                st[t] = a;
            }

            // ---- mask + online softmax (one correction per 64-wide chunk) ----
#pragma unroll
            for (int v = 0; v < 8; ++v) {
                int rq = q0 + wave * 16 + v + 8 * lhi;
                float sv[4];
#pragma unroll
                for (int t = 0; t < 4; ++t) {
                    sv[t] = st[t][v] * 0.125f;
                    if (kc + t * 16 + ln16 > rq) sv[t] = -1e30f;
                }
                float mx = fmaxf(fmaxf(sv[0], sv[1]), fmaxf(sv[2], sv[3]));
                mx = rowred_max(mx);
                float mnew = fmaxf(m8[v], mx);
                float p[4];
#pragma unroll
                for (int t = 0; t < 4; ++t) p[t] = __expf(sv[t] - mnew);
                float rs = rowred_sum((p[0] + p[1]) + (p[2] + p[3]));
                float corr = __expf(m8[v] - mnew);
                l8[v] = l8[v] * corr + rs;
                m8[v] = mnew;
                o[0][v] *= corr; o[1][v] *= corr; o[2][v] *= corr; o[3][v] *= corr;
                int prow = v + 8 * lhi;
#pragma unroll
                for (int t = 0; t < 4; ++t)
                    ldsP[wave][prow * 64 + t * 16 + ln16] = f16b(p[t]);
            }
            asm volatile("" ::: "memory");  // order P store -> P load (same-wave DS in-order)

            // ---- O += P V (two 32-deep k-steps x four d-tiles) ----
#pragma unroll
            for (int kg = 0; kg < 2; ++kg) {
                HFrag pf;
                pf.q[0] = *(const uint4*)(lP + ln16 * 32 + kg * 16 + lhi * 4);
                pf.q[1] = *(const uint4*)(lP + ln16 * 32 + kg * 16 + 8 + lhi * 4);
#pragma unroll
                for (int dt = 0; dt < 4; ++dt) {
                    HFrag vf;
                    int dcol = dt * 16 + ln16;
                    vf.q[0] = *(const uint4*)(lV + dcol * 32 + kg * 16 + lhi * 4);
                    vf.q[1] = *(const uint4*)(lV + dcol * 32 + kg * 16 + 8 + lhi * 4);
                    o[dt] = __builtin_amdgcn_wmma_f32_16x16x32_f16(
                        false, pf.v, false, vf.v, (short)0, o[dt], false, false);
                }
            }
        }

        unsigned short* t0 = kCur; kCur = kNxt; kNxt = t0;
        unsigned short* t1 = vCur; vCur = vNxt; vNxt = t1;
    }

    // ---- epilogue: normalize and write (B,S,H*DH) f32 ----
#pragma unroll
    for (int dt = 0; dt < 4; ++dt)
#pragma unroll
        for (int v = 0; v < 8; ++v) {
            int r = q0 + wave * 16 + v + 8 * lhi;
            float val = o[dt][v] / l8[v];
            Out[(size_t)(b * S_ + r) * (H_ * DH_) + h * DH_ + dt * 16 + ln16] = val;
        }
}

// ---------------------------------------------------------------------------
// Workspace layout (f16, bytes)
// ---------------------------------------------------------------------------
static const size_t OFF_X  = 0;                                    // (B,S,E)
static const size_t OFF_W  = OFF_X  + (size_t)B_ * S_ * E_ * 2;    // swizzled weights
static const size_t OFF_Q  = OFF_W  + (size_t)3 * H_ * E_ * DH_ * 2;
static const size_t OFF_K  = OFF_Q  + (size_t)B_ * H_ * S_ * DH_ * 2;
static const size_t OFF_VT = OFF_K  + (size_t)B_ * H_ * S_ * DH_ * 2;

extern "C" void kernel_launch(void* const* d_in, const int* in_sizes, int n_in,
                              void* d_out, int out_size, void* d_ws, size_t ws_size,
                              hipStream_t stream) {
    const float* X  = (const float*)d_in[0];
    const float* Wq = (const float*)d_in[1];
    const float* Wk = (const float*)d_in[2];
    const float* Wv = (const float*)d_in[3];
    float* out = (float*)d_out;

    char* ws = (char*)d_ws;
    unsigned short* Xh   = (unsigned short*)(ws + OFF_X);
    unsigned short* Wswz = (unsigned short*)(ws + OFF_W);
    unsigned short* Qh   = (unsigned short*)(ws + OFF_Q);
    unsigned short* Kh   = (unsigned short*)(ws + OFF_K);
    unsigned short* Vth  = (unsigned short*)(ws + OFF_VT);

    const int nX = B_ * S_ * E_;
    const int nW = H_ * E_ * DH_;

    cvt_f16_kernel<<<(nX / 4 + 255) / 256, 256, 0, stream>>>(X, Xh, nX / 4);
    wswz_kernel<<<nW / 256, 256, 0, stream>>>(Wq, Wswz);
    wswz_kernel<<<nW / 256, 256, 0, stream>>>(Wk, Wswz + nW);
    wswz_kernel<<<nW / 256, 256, 0, stream>>>(Wv, Wswz + 2 * nW);

    dim3 gp(B_ * H_, S_ / 256, 3);
    qkv_proj_kernel<<<gp, 256, 0, stream>>>(Xh, Wswz, Qh, Kh, Vth);

    dim3 ga(B_ * H_, S_ / 128);
    flash_attn_kernel<<<ga, 256, 0, stream>>>(Qh, Kh, Vth, out);
}
